// SuperPointLoss_64269890617459
// MI455X (gfx1250) — compile-verified, matrix operand order checked by reading, not run on hardware
//
#include <hip/hip_runtime.h>
#include <math.h>

typedef __attribute__((ext_vector_type(16))) _Float16 v16h;
typedef __attribute__((ext_vector_type(8)))  _Float16 v8h;
typedef __attribute__((ext_vector_type(8)))  float    v8f;

#define B_   2
#define H_   480
#define W_   640
#define D_   256
#define Hc_  60
#define Wc_  80
#define N_   (Hc_ * Wc_)       /* 4800 */
#define TILE 96                /* workgroup output tile: 96x96, 2x2 waves of 48x48 */
#define KSTEP 32
#define NKSTEP (D_ / KSTEP)    /* 8 */
#define LSTR 40                /* LDS row stride in halves: 32 + 8 pad, 80B (16B aligned) */

// ---------------------------------------------------------------- zero accum
__global__ void zero_acc_kernel(float* acc) {
    if (threadIdx.x < 16) acc[threadIdx.x] = 0.0f;
}

// ---------------------------------------------------------------- detector loss reduction
__global__ void det_reduce_kernel(const float* __restrict__ pred,
                                  const float* __restrict__ ht,
                                  const float* __restrict__ vm,
                                  float* acc3, float* normAcc, int total) {
    __shared__ float sh[256];
    float pl = 0.f, nl = 0.f, np = 0.f, nm = 0.f;
    for (int i = blockIdx.x * blockDim.x + threadIdx.x; i < total;
         i += gridDim.x * blockDim.x) {
        float p = pred[i], h = ht[i], v = vm[i];
        float pos = (h == 1.0f) ? 1.f : 0.f;
        float neg = (h < 1.0f) ? 1.f : 0.f;
        float omp = 1.f - p;
        pl += logf(p + 1e-7f) * omp * omp * pos * v;
        float nw = 1.f - h; float nw2 = nw * nw;
        nl += logf(1.f - p + 1e-7f) * p * p * (nw2 * nw2) * neg * v;
        np += pos;
        nm += v;
    }
    const int tid = threadIdx.x;
    float vals[4] = {pl, nl, np, nm};
    for (int k = 0; k < 4; ++k) {
        sh[tid] = vals[k];
        __syncthreads();
        for (int off = 128; off > 0; off >>= 1) {
            if (tid < off) sh[tid] += sh[tid + off];
            __syncthreads();
        }
        vals[k] = sh[0];
        __syncthreads();
    }
    if (tid == 0) {
        atomicAdd(&acc3[0], vals[0]);
        atomicAdd(&acc3[1], vals[1]);
        atomicAdd(&acc3[2], vals[2]);
        if (normAcc) atomicAdd(normAcc, vals[3]);
    }
}

// ---------------------------------------------------------------- normalize + fp32 -> f16 panels
__global__ void prep_desc_kernel(const float* __restrict__ src,
                                 _Float16* __restrict__ dst, int BN) {
    int t = blockIdx.x * blockDim.x + threadIdx.x;
    if (t >= BN) return;
    int b = t / N_, n = t - b * N_;
    const float* base = src + (size_t)b * D_ * N_ + n;
    float ss = 0.f;
    for (int c = 0; c < D_; ++c) {
        float x = base[(size_t)c * N_];
        ss += x * x;
    }
    float inv = 1.0f / fmaxf(sqrtf(ss), 1e-12f);
    _Float16* o = dst + (size_t)t * D_;
    for (int c = 0; c < D_; ++c)
        o[c] = (_Float16)(base[(size_t)c * N_] * inv);
}

// ---------------------------------------------------------------- warped coords + coarse masks
__global__ void prep_geom_kernel(const float* __restrict__ Hmat,
                                 const float* __restrict__ vm,
                                 const float* __restrict__ wvm,
                                 float* __restrict__ warp,
                                 float* __restrict__ vmc,
                                 float* __restrict__ wvmc) {
    int t = blockIdx.x * blockDim.x + threadIdx.x;
    if (t >= B_ * N_) return;
    int b = t / N_, n = t - b * N_;
    int i = n / Wc_, j = n - i * Wc_;
    const float* Hm = Hmat + b * 9;
    float px = (float)(j * 8 + 4), py = (float)(i * 8 + 4);
    float xw = Hm[0] * px + Hm[1] * py + Hm[2];
    float yw = Hm[3] * px + Hm[4] * py + Hm[5];
    float zw = Hm[6] * px + Hm[7] * py + Hm[8];
    float invz = 1.0f / zw;
    warp[2 * t]     = yw * invz;   /* row */
    warp[2 * t + 1] = xw * invz;   /* col */

    float sy = (i + 0.5f) * ((float)H_ / Hc_) - 0.5f;
    sy = fminf(fmaxf(sy, 0.f), (float)(H_ - 1));
    int y0 = (int)floorf(sy); int y1 = min(y0 + 1, H_ - 1); float wy = sy - (float)y0;
    float sx = (j + 0.5f) * ((float)W_ / Wc_) - 0.5f;
    sx = fminf(fmaxf(sx, 0.f), (float)(W_ - 1));
    int x0 = (int)floorf(sx); int x1 = min(x0 + 1, W_ - 1); float wx = sx - (float)x0;

    const float* mb = vm + (size_t)b * H_ * W_;
    float a0 = mb[y0 * W_ + x0] * (1.f - wx) + mb[y0 * W_ + x1] * wx;
    float a1 = mb[y1 * W_ + x0] * (1.f - wx) + mb[y1 * W_ + x1] * wx;
    vmc[t] = (a0 * (1.f - wy) + a1 * wy) > 0.5f ? 1.f : 0.f;

    const float* wb = wvm + (size_t)b * H_ * W_;
    float b0 = wb[y0 * W_ + x0] * (1.f - wx) + wb[y0 * W_ + x1] * wx;
    float b1 = wb[y1 * W_ + x0] * (1.f - wx) + wb[y1 * W_ + x1] * wx;
    wvmc[t] = (b0 * (1.f - wy) + b1 * wy) > 0.5f ? 1.f : 0.f;
}

// ---------------------------------------------------------------- gfx1250 async global->LDS copy
__device__ __forceinline__ unsigned lds_off(const void* p) {
    /* LDS aperture: flat addr[31:0] is the LDS address */
    return (unsigned)(uintptr_t)p;
}
__device__ __forceinline__ void async_copy_b128(unsigned ldsAddr, const void* gptr) {
    asm volatile("global_load_async_to_lds_b128 %0, %1, off"
                 :: "v"(ldsAddr), "v"((unsigned long long)(uintptr_t)gptr)
                 : "memory");
}
__device__ __forceinline__ void wait_async0() {
    asm volatile("s_wait_asynccnt 0x0" ::: "memory");
}

/* stage one 96x32-half panel pair (A,B) into LDS via async DMA: 384 chunks each */
__device__ __forceinline__ void stage_async(const _Float16* __restrict__ Ab,
                                            const _Float16* __restrict__ Bb,
                                            _Float16* la, _Float16* lb,
                                            int k0, int tid) {
    #pragma unroll
    for (int c0 = 0; c0 < 3; ++c0) {
        int c = tid + c0 * 128;          /* 0..383 */
        int row = c >> 2, seg = c & 3;
        async_copy_b128(lds_off(&la[row * LSTR + seg * 8]),
                        Ab + (size_t)row * D_ + k0 + seg * 8);
        async_copy_b128(lds_off(&lb[row * LSTR + seg * 8]),
                        Bb + (size_t)row * D_ + k0 + seg * 8);
    }
}

// ---------------------------------------------------------------- fused WMMA GEMM + loss epilogue
__launch_bounds__(128)
__global__ void desc_gemm_kernel(const _Float16* __restrict__ Af,
                                 const _Float16* __restrict__ Bf,
                                 const float* __restrict__ warp,
                                 const float* __restrict__ vmc,
                                 const float* __restrict__ wvmc,
                                 float* __restrict__ lossAcc) {
    __shared__ _Float16 lds[2][2][TILE * LSTR];   /* [buf][A/B][96*40] = 30 KB */
    __shared__ float red[128];

    const int b     = blockIdx.z;
    const int qTile = blockIdx.y * TILE;   /* query rows (d)   -> M */
    const int pTile = blockIdx.x * TILE;   /* key rows (wd)    -> N */
    const int tid   = threadIdx.x;
    const int lane  = tid & 31;
    const int wave  = tid >> 5;
    const int hi    = lane >> 4;           /* 0/1: K-half selector */
    const int l16   = lane & 15;
    const int wq    = (wave >> 1) * 48;    /* wave's 48x48 sub-tile */
    const int wp    = (wave & 1) * 48;

    const _Float16* Ab = Af + ((size_t)b * N_ + qTile) * D_;
    const _Float16* Bb = Bf + ((size_t)b * N_ + pTile) * D_;

    v8f acc[3][3] = {};

    /* prologue: async-fill buffer 0 */
    stage_async(Ab, Bb, &lds[0][0][0], &lds[0][1][0], 0, tid);

    for (int kk = 0; kk < NKSTEP; ++kk) {
        wait_async0();
        __syncthreads();                   /* buf[kk&1] ready; previous reads drained */
        if (kk + 1 < NKSTEP)               /* overlap next panel DMA with WMMA burst */
            stage_async(Ab, Bb, &lds[(kk + 1) & 1][0][0], &lds[(kk + 1) & 1][1][0],
                        (kk + 1) * KSTEP, tid);

        const _Float16* la = &lds[kk & 1][0][0];
        const _Float16* lb = &lds[kk & 1][1][0];

        v16h afrag[3], bfrag[3];
        #pragma unroll
        for (int s = 0; s < 3; ++s) {
            /* A 16x32 f16 layout: lanes 0-15 K[0..7]+K[16..23], lanes 16-31 K[8..15]+K[24..31] */
            int ar = wq + s * 16 + l16;
            v8h lo = *(const v8h*)(&la[ar * LSTR + hi * 8]);
            v8h hv = *(const v8h*)(&la[ar * LSTR + 16 + hi * 8]);
            afrag[s] = __builtin_shufflevector(lo, hv, 0,1,2,3,4,5,6,7,8,9,10,11,12,13,14,15);
            /* B 32x16: column n of wd^T == row n of wd (contiguous), mirrored K-halves */
            int br = wp + s * 16 + l16;
            lo = *(const v8h*)(&lb[br * LSTR + hi * 8]);
            hv = *(const v8h*)(&lb[br * LSTR + 16 + hi * 8]);
            bfrag[s] = __builtin_shufflevector(lo, hv, 0,1,2,3,4,5,6,7,8,9,10,11,12,13,14,15);
        }

        #pragma unroll
        for (int si = 0; si < 3; ++si)
            #pragma unroll
            for (int sj = 0; sj < 3; ++sj)
                acc[si][sj] = __builtin_amdgcn_wmma_f32_16x16x32_f16(
                    false, afrag[si], false, bfrag[sj],
                    (short)0, acc[si][sj], false, false);
    }

    /* fused loss epilogue: C/D layout lane L, VGPR r -> M = r + 8*(L>=16), N = L&15 */
    float lsum = 0.f;
    const float* warpB = warp + (size_t)b * N_ * 2;
    const float* vmcB  = vmc  + (size_t)b * N_;
    const float* wvmcB = wvmc + (size_t)b * N_;

    #pragma unroll
    for (int si = 0; si < 3; ++si) {
        #pragma unroll
        for (int sj = 0; sj < 3; ++sj) {
            int p  = pTile + wp + sj * 16 + l16;
            int pi = p / Wc_, pj = p - pi * Wc_;
            float cy = (float)(pi * 8 + 4), cx = (float)(pj * 8 + 4);
            float wvp = wvmcB[p];
            #pragma unroll
            for (int r = 0; r < 8; ++r) {
                int q = qTile + wq + si * 16 + r + 8 * hi;
                float wy = warpB[2 * q], wx = warpB[2 * q + 1];
                float dy = cy - wy, dx = cx - wx;
                float s  = (dy * dy + dx * dx <= 56.25f) ? 1.f : 0.f;  /* dist <= 7.5 */
                float valid = vmcB[q] * wvp;
                float d   = fmaxf(acc[si][sj][r], 0.f);
                float pd  = fmaxf(0.f, 1.0f - d);
                float nd  = fmaxf(0.f, d - 0.2f);
                lsum += (250.f * s * pd + (1.f - s) * nd) * valid;
            }
        }
    }

    red[tid] = lsum;
    __syncthreads();
    for (int off = 64; off > 0; off >>= 1) {
        if (tid < off) red[tid] += red[tid + off];
        __syncthreads();
    }
    if (tid == 0) atomicAdd(lossAcc, red[0]);
}

// ---------------------------------------------------------------- finalize
__global__ void finalize_kernel(const float* __restrict__ acc, float* __restrict__ out) {
    if (threadIdx.x != 0 || blockIdx.x != 0) return;
    float pl = acc[0], nl = acc[1], np = acc[2];
    float points = (np == 0.f) ? -nl : -(pl + nl) / fmaxf(np, 1.f);
    float wpl = acc[3], wnl = acc[4], wnp = acc[5];
    float wpoints = (wnp == 0.f) ? -wnl : -(wpl + wnl) / fmaxf(wnp, 1.f);
    float desc = acc[6] / acc[7];
    out[0] = points + wpoints + 1e-4f * desc;
    out[1] = points;
    out[2] = wpoints;
    out[3] = desc;
}

// ---------------------------------------------------------------- launch
extern "C" void kernel_launch(void* const* d_in, const int* in_sizes, int n_in,
                              void* d_out, int out_size, void* d_ws, size_t ws_size,
                              hipStream_t stream) {
    (void)in_sizes; (void)n_in; (void)out_size; (void)ws_size;
    const float* prob  = (const float*)d_in[0];
    const float* ht    = (const float*)d_in[1];
    const float* vm    = (const float*)d_in[2];
    const float* wprob = (const float*)d_in[3];
    const float* wht   = (const float*)d_in[4];
    const float* wvm   = (const float*)d_in[5];
    const float* desc  = (const float*)d_in[6];
    const float* wdesc = (const float*)d_in[7];
    const float* Hmat  = (const float*)d_in[8];

    char* ws = (char*)d_ws;
    float* acc = (float*)ws;                       /* 16 floats of accumulators */
    size_t off = 64;
    _Float16* Af = (_Float16*)(ws + off); off += (size_t)B_ * N_ * D_ * sizeof(_Float16);
    _Float16* Bf = (_Float16*)(ws + off); off += (size_t)B_ * N_ * D_ * sizeof(_Float16);
    float* warp  = (float*)(ws + off);    off += (size_t)B_ * N_ * 2 * sizeof(float);
    float* vmc   = (float*)(ws + off);    off += (size_t)B_ * N_ * sizeof(float);
    float* wvmc  = (float*)(ws + off);    off += (size_t)B_ * N_ * sizeof(float);

    zero_acc_kernel<<<1, 32, 0, stream>>>(acc);

    const int total = B_ * H_ * W_;
    det_reduce_kernel<<<512, 256, 0, stream>>>(prob, ht, vm, acc + 0, nullptr, total);
    det_reduce_kernel<<<512, 256, 0, stream>>>(wprob, wht, wvm, acc + 3, acc + 7, total);

    const int bn = B_ * N_;
    prep_desc_kernel<<<(bn + 255) / 256, 256, 0, stream>>>(desc,  Af, bn);
    prep_desc_kernel<<<(bn + 255) / 256, 256, 0, stream>>>(wdesc, Bf, bn);
    prep_geom_kernel<<<(bn + 255) / 256, 256, 0, stream>>>(Hmat, vm, wvm, warp, vmc, wvmc);

    dim3 grid(N_ / TILE, N_ / TILE, B_);           /* 50 x 50 x 2 */
    desc_gemm_kernel<<<grid, 128, 0, stream>>>(Af, Bf, warp, vmc, wvmc, acc + 6);

    finalize_kernel<<<1, 1, 0, stream>>>(acc, (float*)d_out);
}
